// ResidualBlock_87591563034893
// MI455X (gfx1250) — matrix operand order repaired and checked by reading, not compiled, not run on hardware
//
#include <hip/hip_runtime.h>

// Mamba-style residual block, fully fused for MI455X (gfx1250, wave32).
// Bandwidth-bound: ~29 GFLOP vs ~16us HBM floor at 23.3 TB/s -> fuse the
// whole chain, keep scan state in registers, use f16 WMMA (f32 accum) for
// the two K=128 GEMMs, fp32 VALU for softplus/exp/scan.

#define DM    128   // d_model
#define RNK   8     // dt_rank
#define LSEQ  512
#define BATCH 512
#define TT    16    // timesteps per tile (= WMMA M)
#define NTILE (LSEQ / TT)

typedef __attribute__((ext_vector_type(16))) _Float16 v16h;
typedef __attribute__((ext_vector_type(8)))  float    v8f;

union Frag16 { v16h v; unsigned int u[8]; };

__device__ __forceinline__ float siluf(float v)     { return v * (1.f / (1.f + __expf(-v))); }
__device__ __forceinline__ float softplusf(float v) { return v > 20.f ? v : log1pf(__expf(v)); }

__launch_bounds__(128)
__global__ void mamba_block_fused(
    const float* __restrict__ x,
    const float* __restrict__ norm_w,
    const float* __restrict__ in_proj_w,   // (128, 256) row-major
    const float* __restrict__ conv_w,      // (128, 1, 4)
    const float* __restrict__ conv_b,      // (128)
    const float* __restrict__ x_proj_w,    // (128, 12)
    const float* __restrict__ dt_proj_w,   // (8, 128)
    const float* __restrict__ dt_proj_b,   // (128)
    const float* __restrict__ A_log,       // (128, 2)
    const float* __restrict__ D_param,     // (128)
    const float* __restrict__ out_proj_w,  // (128, 128)
    float* __restrict__ out)               // (512, 512, 128)
{
    // Weights resident in LDS, transposed [N][K] so WMMA B K-pairs are
    // contiguous; +4 f16 row pad breaks 256B-stride bank conflicts.
    __shared__ _Float16 sWi[2*DM][DM+4];   // in_proj^T  (f16)   67.5 KB
    __shared__ _Float16 sWo[DM][DM+4];     // out_proj^T (f16)   33.8 KB
    __shared__ float    sWx[DM][12];       // x_proj (f32)
    __shared__ float    sWdt[RNK][DM];     // dt_proj (f32)
    __shared__ float    sNw[DM];
    __shared__ _Float16 sA16[TT][DM+8];    // normalized SSM-path tokens (WMMA A)
    __shared__ _Float16 sY16[TT][DM+8];    // gated scan output (WMMA A)
    __shared__ float    sXr[TT][2*DM];     // in_proj result: [0:128)=xs, [128:256)=res
    __shared__ float    sRes[TT][DM];      // normalized residual-path tokens
    __shared__ float    sXdbl[TT][12];     // x_proj result: delta_r | B | C

    const int tid   = threadIdx.x;
    const int b     = blockIdx.x;
    const int wv    = tid >> 5;            // wave 0..3
    const int lane  = tid & 31;
    const bool lolane = lane < 16;
    const int mrow  = lane & 15;
    const int mbase = lolane ? 0 : 8;

    // ---- one-time weight staging ----
    for (int i = tid; i < 2*DM*DM; i += 128) { int c0 = i >> 8; int j = i & 255; sWi[j][c0] = (_Float16)in_proj_w[i]; }
    for (int i = tid; i < DM*DM;   i += 128) { int c0 = i >> 7; int j = i & 127; sWo[j][c0] = (_Float16)out_proj_w[i]; }
    for (int i = tid; i < DM*12;   i += 128) { sWx[i/12][i%12] = x_proj_w[i]; }
    for (int i = tid; i < RNK*DM;  i += 128) { sWdt[i>>7][i&127] = dt_proj_w[i]; }
    if (tid < DM) sNw[tid] = norm_w[tid];

    // per-channel parameters (thread == channel for VALU stages)
    const int c = tid;
    const float cw0 = conv_w[c*4+0], cw1 = conv_w[c*4+1], cw2 = conv_w[c*4+2], cw3 = conv_w[c*4+3];
    const float cb  = conv_b[c];
    const float db  = dt_proj_b[c];
    const float A0  = -__expf(A_log[c*2+0]);
    const float A1  = -__expf(A_log[c*2+1]);
    const float Dp  = D_param[c];

    // register-resident recurrent state (carried across the whole sequence)
    float h1 = 0.f, h2 = 0.f, h3 = 0.f;   // conv history xs_pre[l-1..l-3]
    float s0 = 0.f, s1 = 0.f;             // SSM state (n=2)

    const int tg = tid >> 3;              // token within tile (0..15)
    const int lg = tid & 7;               // 8 lanes cooperate per token

    for (int tile = 0; tile < NTILE; ++tile) {
        const int l0 = tile * TT;
        __syncthreads();   // protects sA16/sRes vs previous tile's stage F

        // ---- Stage A: load + RMSNorm both token streams ----
        {
            // SSM path: h[b,l] = norm(x[l, b, :])  (strided column of x)
            const float* ph = x + ((size_t)(l0 + tg)*BATCH + b)*DM + lg*16;
            float v[16]; float ss = 0.f;
            #pragma unroll
            for (int i = 0; i < 16; ++i) { v[i] = ph[i]; ss += v[i]*v[i]; }
            ss += __shfl_xor(ss, 1, 8); ss += __shfl_xor(ss, 2, 8); ss += __shfl_xor(ss, 4, 8);
            const float rs = rsqrtf(ss * (1.f/128.f) + 1e-5f);
            #pragma unroll
            for (int i = 0; i < 16; ++i)
                sA16[tg][lg*16 + i] = (_Float16)(v[i] * rs * sNw[lg*16 + i]);

            // residual path: x1[b,l] = norm(x[b, l, :]) (contiguous, L2-hot)
            const float* pr = x + ((size_t)b*LSEQ + (l0 + tg))*DM + lg*16;
            float w[16]; float ss2 = 0.f;
            #pragma unroll
            for (int i = 0; i < 16; ++i) { w[i] = pr[i]; ss2 += w[i]*w[i]; }
            ss2 += __shfl_xor(ss2, 1, 8); ss2 += __shfl_xor(ss2, 2, 8); ss2 += __shfl_xor(ss2, 4, 8);
            const float rs2 = rsqrtf(ss2 * (1.f/128.f) + 1e-5f);
            #pragma unroll
            for (int i = 0; i < 16; ++i)
                sRes[tg][lg*16 + i] = w[i] * rs2 * sNw[lg*16 + i];

            if (tile + 1 < NTILE)  // hide the strided column loads of the next tile
                __builtin_prefetch(x + ((size_t)(l0 + TT + tg)*BATCH + b)*DM + lg*16, 0, 0);
        }
        __syncthreads();

        // ---- Stage B: in_proj  (16x128 @ 128x256) via v_wmma_f32_16x16x32_f16 ----
        for (int nt = wv; nt < 16; nt += 4) {
            v8f acc = {0.f,0.f,0.f,0.f,0.f,0.f,0.f,0.f};
            #pragma unroll
            for (int ks = 0; ks < 4; ++ks) {
                Frag16 af, bf;
                // A 16x32 f16 layout (ISA 7.12.2): lanes<16 K in {0..7,16..23}, lanes>=16 +8
                const int ka = ks*32 + (lolane ? 0 : 8);
                #pragma unroll
                for (int i = 0; i < 8; ++i) {
                    const int ko = ka + 2*i + ((i < 4) ? 0 : 8);
                    af.u[i] = *(const unsigned int*)&sA16[mrow][ko];
                }
                // B 32x16: lane = column N; lanes<16 K in {0..7,16..23}, lanes>=16 +8
                const int n  = nt*16 + mrow;
                const int kb = ks*32 + (lolane ? 0 : 8);
                #pragma unroll
                for (int i = 0; i < 8; ++i)
                    bf.u[i] = *(const unsigned int*)&sWi[n][kb + 2*i + ((i < 4) ? 0 : 8)];
                acc = __builtin_amdgcn_wmma_f32_16x16x32_f16(
                          false, af.v, false, bf.v, (short)0, acc, false, false);
            }
            const int n = nt*16 + mrow;
            #pragma unroll
            for (int r = 0; r < 8; ++r) sXr[mbase + r][n] = acc[r];
        }
        __syncthreads();

        // ---- Stage C: depthwise causal conv(k=4) + bias + SiLU (per channel) ----
        #pragma unroll
        for (int t = 0; t < TT; ++t) {
            const float xp = sXr[t][c];
            const float xc = cb + cw0*h3 + cw1*h2 + cw2*h1 + cw3*xp;
            h3 = h2; h2 = h1; h1 = xp;
            sXr[t][c] = siluf(xc);     // overwrite xs in place (column-private)
        }
        __syncthreads();

        // ---- Stage D: x_proj (16x128 @ 128x12), fp32 VALU ----
        for (int p = tid; p < TT*12; p += 128) {
            const int t = p / 12, j = p % 12;
            float s = 0.f;
            for (int cc = 0; cc < DM; ++cc) s += sXr[t][cc] * sWx[cc][j];
            sXdbl[t][j] = s;
        }
        __syncthreads();

        // ---- Stage E: dt_proj + softplus + selective scan + gating (per channel) ----
        #pragma unroll 1
        for (int t = 0; t < TT; ++t) {
            float dt = db;
            #pragma unroll
            for (int r = 0; r < RNK; ++r) dt += sXdbl[t][r] * sWdt[r][c];
            dt = softplusf(dt);
            const float Bt0 = sXdbl[t][8],  Bt1 = sXdbl[t][9];
            const float Ct0 = sXdbl[t][10], Ct1 = sXdbl[t][11];
            s0 = __expf(dt * A0) * s0 + dt * Bt0;    // deltaB = delta*B (per reference)
            s1 = __expf(dt * A1) * s1 + dt * Bt1;
            float y = s0*Ct0 + s1*Ct1 + sXr[t][c]*Dp;
            y *= siluf(sXr[t][c + DM]);              // gate with res half
            sY16[t][c] = (_Float16)y;
        }
        __syncthreads();

        // ---- Stage F: out_proj (16x128 @ 128x128) WMMA + residual + store ----
        for (int nt = wv; nt < 8; nt += 4) {
            v8f acc = {0.f,0.f,0.f,0.f,0.f,0.f,0.f,0.f};
            #pragma unroll
            for (int ks = 0; ks < 4; ++ks) {
                Frag16 af, bf;
                const int ka = ks*32 + (lolane ? 0 : 8);
                #pragma unroll
                for (int i = 0; i < 8; ++i) {
                    const int ko = ka + 2*i + ((i < 4) ? 0 : 8);
                    af.u[i] = *(const unsigned int*)&sY16[mrow][ko];
                }
                const int n  = nt*16 + mrow;
                const int kb = ks*32 + (lolane ? 0 : 8);
                #pragma unroll
                for (int i = 0; i < 8; ++i)
                    bf.u[i] = *(const unsigned int*)&sWo[n][kb + 2*i + ((i < 4) ? 0 : 8)];
                acc = __builtin_amdgcn_wmma_f32_16x16x32_f16(
                          false, af.v, false, bf.v, (short)0, acc, false, false);
            }
            // D-frag: lane -> column n, VGPR r -> row mbase+r; fuse residual add.
            const int n = nt*16 + mrow;
            #pragma unroll
            for (int r = 0; r < 8; ++r) {
                const int t = mbase + r;
                out[((size_t)b*LSEQ + (l0 + t))*DM + n] = acc[r] + sRes[t][n];
            }
        }
    }
}

extern "C" void kernel_launch(void* const* d_in, const int* in_sizes, int n_in,
                              void* d_out, int out_size, void* d_ws, size_t ws_size,
                              hipStream_t stream) {
    const float* x          = (const float*)d_in[0];
    const float* norm_w     = (const float*)d_in[1];
    const float* in_proj_w  = (const float*)d_in[2];
    const float* conv_w     = (const float*)d_in[3];
    const float* conv_b     = (const float*)d_in[4];
    const float* x_proj_w   = (const float*)d_in[5];
    const float* dt_proj_w  = (const float*)d_in[6];
    const float* dt_proj_b  = (const float*)d_in[7];
    const float* A_log      = (const float*)d_in[8];
    const float* D_param    = (const float*)d_in[9];
    const float* out_proj_w = (const float*)d_in[10];
    float* outp = (float*)d_out;

    // one workgroup per batch row; the scan serializes along the sequence,
    // so each block walks its full sequence with state in registers.
    mamba_block_fused<<<dim3(BATCH), dim3(128), 0, stream>>>(
        x, norm_w, in_proj_w, conv_w, conv_b, x_proj_w, dt_proj_w, dt_proj_b,
        A_log, D_param, out_proj_w, outp);
}
